// Transformer_42314017800987
// MI455X (gfx1250) — matrix operand order repaired and checked by reading, not compiled, not run on hardware
//
#include <hip/hip_runtime.h>
#include <hip/hip_bf16.h>

#define DEVI __device__ __forceinline__

typedef _Float16 v16h __attribute__((ext_vector_type(16)));
typedef _Float16 h8   __attribute__((ext_vector_type(8)));
typedef float    v8f  __attribute__((ext_vector_type(8)));
typedef unsigned int u32x4 __attribute__((ext_vector_type(4)));
typedef int          i32x4 __attribute__((ext_vector_type(4)));
typedef int          i32x8 __attribute__((ext_vector_type(8)));

// ---- problem constants ----
#define NTOK   12544   // 64 * 196
#define HID    768
#define LSEQ   196
#define NHEAD  24
#define HD     32
#define FFNDIM 3072
#define LPAD   224     // 196 padded to 7*32
#define INV_SQRT_HD 0.17677669529663687f
#define DROP_RATE 0.2f

#if __has_builtin(__builtin_amdgcn_tensor_load_to_lds)
#define HAS_TDM 1
#else
#define HAS_TDM 0
#endif

DEVI int imin(int a, int b) { return a < b ? a : b; }

DEVI v8f wmma32(v16h a, v16h b, v8f c) {
    // D = A(16x32 f16) * B(32x16 f16) + C(f32)
    return __builtin_amdgcn_wmma_f32_16x16x32_f16(false, a, false, b, (short)0, c, false, false);
}

// A-fragment (16x32, f16): lane holds row (lane&15); lanes 0-15: K {0..7,16..23}, lanes 16-31: K {8..15,24..31}
DEVI v16h ld_afrag(const _Float16* rowp, int lane) {
    const int kb = (lane < 16) ? 0 : 8;
    h8 lo = *(const h8*)(rowp + kb);
    h8 hi = *(const h8*)(rowp + kb + 16);
    v16h a;
#pragma unroll
    for (int i = 0; i < 8; ++i) { a[i] = lo[i]; a[i + 8] = hi[i]; }
    return a;
}

// B-fragment (32x16, f16): lane holds col (lane&15); lanes 0-15: K {0..7,16..23}, lanes 16-31: K {8..15,24..31}.
DEVI v16h ld_bfrag(const _Float16* colp, int lane) {
    const int kb = (lane < 16) ? 0 : 8;
    h8 lo = *(const h8*)(colp + kb);
    h8 hi = *(const h8*)(colp + kb + 16);
    v16h b;
#pragma unroll
    for (int i = 0; i < 8; ++i) { b[i] = lo[i]; b[i + 8] = hi[i]; }
    return b;
}

#if HAS_TDM
// Issue a TDM 2D tile load: tile (tile_x elems wide, tile_y rows) from a row-major
// tensor with row stride stride_x (elements, 2-byte data) into LDS at lds_addr.
// D# layout per CDNA5 ISA ch.8 (group0: control/lds/global/type, group1: dims/strides).
// 6-arg builtin form (clang-23 / therock headers): (g0, g1, g2, g3, g4, cpol).
DEVI void tdm_load_2d(unsigned lds_addr, const void* gaddr,
                      unsigned tile_x, unsigned tile_y,
                      unsigned tensor_x, unsigned tensor_y,
                      unsigned stride_x) {
    unsigned long long ga = (unsigned long long)gaddr;
    u32x4 g0;
    g0[0] = 1u;                                        // count=1, user descriptor
    g0[1] = lds_addr;                                  // LDS byte address
    g0[2] = (unsigned)(ga & 0xFFFFFFFFu);              // global_addr[31:0]
    g0[3] = (unsigned)((ga >> 32) & 0x01FFFFFFu)       // global_addr[56:32]
            | (2u << 30);                              // type=2 ("image")
    i32x8 g1;
    g1[0] = (int)(1u << 16);                           // wg_mask=0, data_size=1 (2 bytes)
    g1[1] = (int)((tensor_x & 0xFFFFu) << 16);         // abar_addr=0, tensor_dim0[15:0]
    g1[2] = (int)((tensor_x >> 16) | ((tensor_y & 0xFFFFu) << 16));  // dim0[31:16], dim1[15:0]
    g1[3] = (int)((tensor_y >> 16) | (tile_x << 16));  // dim1[31:16], tile_dim0
    g1[4] = (int)(tile_y & 0xFFFFu);                   // tile_dim1, tile_dim2=0
    g1[5] = (int)stride_x;                             // tensor_dim0_stride[31:0]
    g1[6] = 0;                                         // stride0[47:32]=0, stride1 lo=0
    g1[7] = 0;
    i32x4 z4; z4[0] = 0; z4[1] = 0; z4[2] = 0; z4[3] = 0;
    i32x8 z8; 
#pragma unroll
    for (int i = 0; i < 8; ++i) z8[i] = 0;
    __builtin_amdgcn_tensor_load_to_lds(g0, g1, z4, z4, z8, 0);
}
#endif

// ---------------- f32 -> f16 weight conversion ----------------
__global__ void cvt_f32_f16(const float* __restrict__ src, _Float16* __restrict__ dst, int n) {
    int i = blockIdx.x * 256 + threadIdx.x;
    if (i < n) dst[i] = (_Float16)src[i];
}

// ---------------- LayerNorm (one wave per token) ----------------
__global__ __launch_bounds__(256) void ln_kernel(const float* __restrict__ x,
                                                 const float* __restrict__ g,
                                                 const float* __restrict__ b,
                                                 _Float16* __restrict__ out) {
    int wave = threadIdx.x >> 5, lane = threadIdx.x & 31;
    int tok = blockIdx.x * 8 + wave;
    const float* row = x + (size_t)tok * HID;
    float v[24];
    float s = 0.f;
#pragma unroll
    for (int i = 0; i < 24; ++i) { v[i] = row[i * 32 + lane]; s += v[i]; }
#pragma unroll
    for (int o = 16; o; o >>= 1) s += __shfl_xor(s, o, 32);
    float mu = s * (1.f / (float)HID);
    float vs = 0.f;
#pragma unroll
    for (int i = 0; i < 24; ++i) { float d = v[i] - mu; vs += d * d; }
#pragma unroll
    for (int o = 16; o; o >>= 1) vs += __shfl_xor(vs, o, 32);
    float rstd = rsqrtf(vs * (1.f / (float)HID) + 1e-5f);
    _Float16* orow = out + (size_t)tok * HID;
#pragma unroll
    for (int i = 0; i < 24; ++i) {
        int c = i * 32 + lane;
        orow[c] = (_Float16)((v[i] - mu) * rstd * g[c] + b[c]);
    }
}

// ---------------- QKV projection: per-head 32x32 weights, shared across heads ----------------
__global__ __launch_bounds__(128) void qkv_kernel(const _Float16* __restrict__ xn,
                                                  const float* __restrict__ wq,
                                                  const float* __restrict__ wk,
                                                  const float* __restrict__ wv,
                                                  _Float16* __restrict__ qo,
                                                  _Float16* __restrict__ ko,
                                                  _Float16* __restrict__ vo) {
    __shared__ _Float16 s_w[3][HD * HD];
    int tid = threadIdx.x;
    for (int i = tid; i < HD * HD; i += 128) {
        s_w[0][i] = (_Float16)wq[i];
        s_w[1][i] = (_Float16)wk[i];
        s_w[2][i] = (_Float16)wv[i];
    }
    __syncthreads();

    int lane = tid & 31, w = tid >> 5;
    int head = blockIdx.y * 4 + w;
    int tok0 = blockIdx.x * 16;
    int nl = lane & 15, half8 = (lane >> 4) * 8;

    const _Float16* ap = xn + (size_t)(tok0 + nl) * HID + head * HD;
    v16h a = ld_afrag(ap, lane);

    _Float16* outs[3] = { qo, ko, vo };
#pragma unroll
    for (int wi = 0; wi < 3; ++wi) {
#pragma unroll
        for (int nh = 0; nh < 2; ++nh) {
            v16h b = ld_bfrag(&s_w[wi][(nh * 16 + nl) * HD], lane);
            v8f c = {};
            c = wmma32(a, b, c);
#pragma unroll
            for (int r = 0; r < 8; ++r) {
                int tok = tok0 + r + half8;
                outs[wi][(size_t)tok * HID + head * HD + nh * 16 + nl] = (_Float16)c[r];
            }
        }
    }
}

// ---------------- Attention: one block per (head, batch) ----------------
__global__ __launch_bounds__(128) void attn_kernel(const _Float16* __restrict__ q,
                                                   const _Float16* __restrict__ k,
                                                   const _Float16* __restrict__ v,
                                                   const float* __restrict__ rel_bias,
                                                   _Float16* __restrict__ o) {
    __shared__ float    s_bias[784];
    __shared__ _Float16 s_k[LPAD * HD];              // keys  (key, d), rows >=196 zero
    __shared__ _Float16 s_vt[HD * LPAD];             // V^T   (e, key), cols >=196 zero
    __shared__ _Float16 s_p[4 * 16 * LPAD];          // per-wave probs (row, key) f16

    int head = blockIdx.x, n = blockIdx.y;
    int tid = threadIdx.x, lane = tid & 31, w = tid >> 5;
    size_t base = ((size_t)n * LSEQ) * HID + head * HD;

    for (int i = tid; i < 784; i += 128) s_bias[i] = rel_bias[i];

#if HAS_TDM
    // TDM: DMA the 196x32 key tile straight into LDS (rows packed, stride HID).
    if (w == 0) {
        tdm_load_2d((unsigned)(unsigned long long)(const void*)&s_k[0], &k[base],
                    HD, LSEQ, HID, NTOK, HID);
        __builtin_amdgcn_s_wait_tensorcnt(0);
    }
    // zero the pad rows 196..223 (disjoint from the DMA target region)
    for (int idx = tid; idx < ((LPAD - LSEQ) * HD) / 8; idx += 128) {
        h8 z; for (int i = 0; i < 8; ++i) z[i] = (_Float16)0;
        *(h8*)&s_k[LSEQ * HD + idx * 8] = z;
    }
#else
    for (int idx = tid; idx < (LPAD * HD) / 8; idx += 128) {
        int key = idx >> 2, co = (idx & 3) * 8;
        h8 val;
        if (key < LSEQ) val = *(const h8*)&k[base + (size_t)key * HID + co];
        else { for (int i = 0; i < 8; ++i) val[i] = (_Float16)0; }
        *(h8*)&s_k[key * HD + co] = val;
    }
#endif
    // stage V^T (scalar, strided gather)
    for (int idx = tid; idx < HD * LPAD; idx += 128) {
        int e = idx / LPAD, key = idx % LPAD;
        s_vt[idx] = (key < LSEQ) ? v[base + (size_t)key * HID + e] : (_Float16)0;
    }
    // zero probs (padded key columns must be 0 for the PV GEMM)
    for (int idx = tid; idx < (4 * 16 * LPAD) / 8; idx += 128) {
        h8 z; for (int i = 0; i < 8; ++i) z[i] = (_Float16)0;
        *(h8*)&s_p[idx * 8] = z;
    }
    __syncthreads();

    int nl = lane & 15, half8 = (lane >> 4) * 8;
    _Float16* myp = &s_p[w * 16 * LPAD];

    for (int qt = w; qt < 13; qt += 4) {
        int qbase = qt * 16;
        int tokc = imin(qbase + nl, LSEQ - 1);
        v16h aq = ld_afrag(&q[base + (size_t)tokc * HID], lane);

        float sc[13][8];
#pragma unroll
        for (int t = 0; t < 13; ++t) {
            v16h bk = ld_bfrag(&s_k[(t * 16 + nl) * HD], lane);
            v8f c = {};
            c = wmma32(aq, bk, c);
            int kidx = t * 16 + nl;
            int kc = imin(kidx, LSEQ - 1);
            int ip = kc / 14, jp = kc % 14;
#pragma unroll
            for (int r = 0; r < 8; ++r) {
                int qi = imin(qbase + r + half8, LSEQ - 1);
                int i = qi / 14, j = qi % 14;
                float bb = s_bias[(i - ip + 14) * 28 + (j - jp + 14)];
                sc[t][r] = (kidx < LSEQ) ? (c[r] + bb) * INV_SQRT_HD : -1e30f;
            }
        }

        float mx[8], sm[8];
#pragma unroll
        for (int r = 0; r < 8; ++r) {
            float m2 = sc[0][r];
#pragma unroll
            for (int t = 1; t < 13; ++t) m2 = fmaxf(m2, sc[t][r]);
#pragma unroll
            for (int off = 1; off < 16; off <<= 1) m2 = fmaxf(m2, __shfl_xor(m2, off, 32));
            mx[r] = m2;
        }
#pragma unroll
        for (int r = 0; r < 8; ++r) {
            float s2 = 0.f;
            int m = r + half8;
#pragma unroll
            for (int t = 0; t < 13; ++t) {
                float e = (sc[t][r] > -1e29f) ? __expf(sc[t][r] - mx[r]) : 0.f;
                myp[m * LPAD + t * 16 + nl] = (_Float16)e;
                s2 += e;
            }
#pragma unroll
            for (int off = 1; off < 16; off <<= 1) s2 += __shfl_xor(s2, off, 32);
            sm[r] = s2;
        }

        v8f o0 = {}, o1 = {};
#pragma unroll
        for (int c2 = 0; c2 < 7; ++c2) {
            v16h ap = ld_afrag(&myp[nl * LPAD + c2 * 32], lane);
            v16h bv0 = ld_bfrag(&s_vt[(0 * 16 + nl) * LPAD + c2 * 32], lane);
            v16h bv1 = ld_bfrag(&s_vt[(1 * 16 + nl) * LPAD + c2 * 32], lane);
            o0 = wmma32(ap, bv0, o0);
            o1 = wmma32(ap, bv1, o1);
        }
#pragma unroll
        for (int r = 0; r < 8; ++r) {
            int qi = qbase + r + half8;
            if (qi < LSEQ) {
                float inv = 1.f / sm[r];
                o[base + (size_t)qi * HID + nl]      = (_Float16)(o0[r] * inv);
                o[base + (size_t)qi * HID + 16 + nl] = (_Float16)(o1[r] * inv);
            }
        }
    }
}

// ---------------- Tiled GEMM: out = A(MxK,f16) @ W(NxK,f16)^T, 3 epilogues ----------------
// Stage: BK=64; A/B tiles DMA'd into LDS by the Tensor Data Mover (wave 0 issues,
// TENSORcnt-waited, barrier-released); 8 WMMAs per wave per stage.
// mode 0: outf = (resid + acc) * scale
// mode 1: outh = gelu(acc + bias)
// mode 2: outf = (resid + acc + bias) * scale
__global__ __launch_bounds__(256) void gemm16(const _Float16* __restrict__ A,
                                              const _Float16* __restrict__ W,
                                              const float* __restrict__ bias,
                                              const float* __restrict__ resid,
                                              float* __restrict__ outf,
                                              _Float16* __restrict__ outh,
                                              int M, int N, int K, int mode, float scale) {
    __shared__ _Float16 As[128 * 64];
    __shared__ _Float16 Bs[64 * 64];
    int tid = threadIdx.x, lane = tid & 31, w = tid >> 5;
    int wm = w >> 1, wn = w & 1;
    int m0 = blockIdx.x * 128, n0 = blockIdx.y * 64;
    int nl = lane & 15, half8 = (lane >> 4) * 8;

    v8f acc[2][2] = { { {}, {} }, { {}, {} } };

    for (int k0 = 0; k0 < K; k0 += 64) {
        __syncthreads();
#if HAS_TDM
        if (w == 0) {
            tdm_load_2d((unsigned)(unsigned long long)(const void*)&As[0],
                        &A[(size_t)m0 * K + k0], 64, 128, (unsigned)K, (unsigned)M, (unsigned)K);
            tdm_load_2d((unsigned)(unsigned long long)(const void*)&Bs[0],
                        &W[(size_t)n0 * K + k0], 64, 64, (unsigned)K, (unsigned)N, (unsigned)K);
            __builtin_amdgcn_s_wait_tensorcnt(0);
        }
        if (k0 + 64 < K)                              // L2 prefetch of next A stage
            __builtin_prefetch(&A[(size_t)(m0 + (tid >> 1)) * K + k0 + 64], 0, 0);
#else
#pragma unroll
        for (int c = 0; c < 4; ++c) {                 // 128x64 A tile, 16B per store
            int idx = tid + c * 256;
            int r = idx >> 3, co = (idx & 7) * 8;
            *(h8*)&As[r * 64 + co] = *(const h8*)&A[(size_t)(m0 + r) * K + k0 + co];
        }
#pragma unroll
        for (int c = 0; c < 2; ++c) {                 // 64x64 B tile
            int idx = tid + c * 256;
            int r = idx >> 3, co = (idx & 7) * 8;
            *(h8*)&Bs[r * 64 + co] = *(const h8*)&W[(size_t)(n0 + r) * K + k0 + co];
        }
        if (k0 + 64 < K)
            __builtin_prefetch(&A[(size_t)(m0 + (tid >> 1)) * K + k0 + 64], 0, 0);
#endif
        __syncthreads();

#pragma unroll
        for (int kc = 0; kc < 2; ++kc) {
            v16h a[2], b[2];
#pragma unroll
            for (int i = 0; i < 2; ++i)
                a[i] = ld_afrag(&As[(wm * 32 + i * 16 + nl) * 64 + kc * 32], lane);
#pragma unroll
            for (int j = 0; j < 2; ++j)
                b[j] = ld_bfrag(&Bs[(wn * 32 + j * 16 + nl) * 64 + kc * 32], lane);
#pragma unroll
            for (int i = 0; i < 2; ++i)
#pragma unroll
                for (int j = 0; j < 2; ++j)
                    acc[i][j] = wmma32(a[i], b[j], acc[i][j]);
        }
    }

#pragma unroll
    for (int i = 0; i < 2; ++i)
#pragma unroll
        for (int j = 0; j < 2; ++j)
#pragma unroll
            for (int r = 0; r < 8; ++r) {
                int m = m0 + wm * 32 + i * 16 + r + half8;
                int n = n0 + wn * 32 + j * 16 + nl;
                float val = acc[i][j][r];
                size_t oidx = (size_t)m * N + n;
                if (mode == 0) {
                    outf[oidx] = (resid[oidx] + val) * scale;
                } else if (mode == 1) {
                    float t = val + bias[n];
                    outh[oidx] = (_Float16)(0.5f * t * (1.f + erff(t * 0.70710678118654752f)));
                } else {
                    float t = resid[(size_t)m * HID + n] + val + bias[n];
                    outf[oidx] = t * scale;
                }
            }
}

// ---------------- workspace layout (bytes) ----------------
#define X1_OFF   ((size_t)0)
#define XN_OFF   (X1_OFF + (size_t)NTOK * HID * 4)
#define Q_OFF    (XN_OFF + (size_t)NTOK * HID * 2)
#define K_OFF    (Q_OFF  + (size_t)NTOK * HID * 2)
#define V_OFF    (K_OFF  + (size_t)NTOK * HID * 2)
#define O_OFF    (V_OFF  + (size_t)NTOK * HID * 2)
#define H_OFF    Q_OFF                               // FFN hidden reuses q/k/v/o region
#define WO_OFF   (O_OFF  + (size_t)NTOK * HID * 2)
#define W1_OFF   (WO_OFF + (size_t)HID * HID * 2)
#define W2_OFF   (W1_OFF + (size_t)FFNDIM * HID * 2)

extern "C" void kernel_launch(void* const* d_in, const int* in_sizes, int n_in,
                              void* d_out, int out_size, void* d_ws, size_t ws_size,
                              hipStream_t stream) {
    const float* x        = (const float*)d_in[0];
    const float* rel_bias = (const float*)d_in[1];
    const float* wq       = (const float*)d_in[2];
    const float* wk       = (const float*)d_in[3];
    const float* wv       = (const float*)d_in[4];
    const float* w_out    = (const float*)d_in[5];
    const float* ln1_g    = (const float*)d_in[6];
    const float* ln1_b    = (const float*)d_in[7];
    const float* ln2_g    = (const float*)d_in[8];
    const float* ln2_b    = (const float*)d_in[9];
    const float* w1       = (const float*)d_in[10];
    const float* b1       = (const float*)d_in[11];
    const float* w2       = (const float*)d_in[12];
    const float* b2       = (const float*)d_in[13];

    char* ws = (char*)d_ws;
    float*     x1  = (float*)(ws + X1_OFF);
    _Float16*  xn  = (_Float16*)(ws + XN_OFF);
    _Float16*  qb  = (_Float16*)(ws + Q_OFF);
    _Float16*  kb  = (_Float16*)(ws + K_OFF);
    _Float16*  vb  = (_Float16*)(ws + V_OFF);
    _Float16*  ob  = (_Float16*)(ws + O_OFF);
    _Float16*  hb  = (_Float16*)(ws + H_OFF);
    _Float16*  woh = (_Float16*)(ws + WO_OFF);
    _Float16*  w1h = (_Float16*)(ws + W1_OFF);
    _Float16*  w2h = (_Float16*)(ws + W2_OFF);

    cvt_f32_f16<<<(HID * HID + 255) / 256, 256, 0, stream>>>(w_out, woh, HID * HID);
    cvt_f32_f16<<<(FFNDIM * HID + 255) / 256, 256, 0, stream>>>(w1, w1h, FFNDIM * HID);
    cvt_f32_f16<<<(FFNDIM * HID + 255) / 256, 256, 0, stream>>>(w2, w2h, FFNDIM * HID);

    // xn = LN1(x)
    ln_kernel<<<NTOK / 8, 256, 0, stream>>>(x, ln1_g, ln1_b, xn);
    // q,k,v
    qkv_kernel<<<dim3(NTOK / 16, NHEAD / 4), 128, 0, stream>>>(xn, wq, wk, wv, qb, kb, vb);
    // attention -> ob
    attn_kernel<<<dim3(NHEAD, 64), 128, 0, stream>>>(qb, kb, vb, rel_bias, ob);
    // x1 = (x + ob @ w_out^T) * RATE
    gemm16<<<dim3(NTOK / 128, HID / 64), 256, 0, stream>>>(ob, woh, nullptr, x, x1, nullptr,
                                                           NTOK, HID, HID, 0, DROP_RATE);
    // xn = LN2(x1)
    ln_kernel<<<NTOK / 8, 256, 0, stream>>>(x1, ln2_g, ln2_b, xn);
    // h = gelu(xn @ w1^T + b1)
    gemm16<<<dim3(NTOK / 128, FFNDIM / 64), 256, 0, stream>>>(xn, w1h, b1, nullptr, nullptr, hb,
                                                              NTOK, FFNDIM, HID, 1, 1.f);
    // out = (x1 + h @ w2^T + b2) * RATE
    gemm16<<<dim3(NTOK / 128, HID / 64), 256, 0, stream>>>(hb, w2h, b2, x1, (float*)d_out, nullptr,
                                                           NTOK, HID, FFNDIM, 2, DROP_RATE);
}
